// LSTMCell_62491774156950
// MI455X (gfx1250) — matrix operand order repaired and checked by reading, not compiled, not run on hardware
//
#include <hip/hip_runtime.h>

typedef __attribute__((ext_vector_type(16))) __bf16 v16bf;
typedef __attribute__((ext_vector_type(8)))  float  v8f;
typedef __attribute__((ext_vector_type(4)))  int    v4i;

#define BDIM 8192
#define HDIM 1024
#define KTOT 2048

#define MT 256      // rows per workgroup tile (2 row-blocks per wave)
#define NT 32       // cols per gate per workgroup tile
#define KT 32       // k-step (WMMA bf16 K)
#define LDA 40      // LDS stride (bf16 elems), padded vs 32 for bank spread
#define LDB 40

#define LOG2E 1.4426950408889634f

// Async copy global->LDS (gfx1250 GLOBAL_LOAD_ASYNC_TO_LDS_B128) with sync fallback.
#ifndef USE_ASYNC
#if defined(__HIP_DEVICE_COMPILE__) && defined(__has_builtin)
#if __has_builtin(__builtin_amdgcn_global_load_async_to_lds_b128)
#define USE_ASYNC 1
#else
#define USE_ASYNC 0
#endif
#else
#define USE_ASYNC 0
#endif
#endif

typedef __attribute__((address_space(1))) v4i* gptr_v4i;
typedef __attribute__((address_space(3))) v4i* lptr_v4i;

union FragU {
    v16bf v;
    uint4 q[2];
};

__device__ __forceinline__ unsigned short f2bf(float f) {
    unsigned u = __builtin_bit_cast(unsigned, f);
    unsigned r = 0x7FFFu + ((u >> 16) & 1u);   // round-to-nearest-even
    return (unsigned short)((u + r) >> 16);
}

// Fast activations on the hardware TRANS pipe (v_exp_f32 / v_rcp_f32),
// saturation-safe at +-inf.
__device__ __forceinline__ float fast_sigmoid(float x) {
    return __builtin_amdgcn_rcpf(1.0f + __builtin_amdgcn_exp2f(-x * LOG2E));
}
__device__ __forceinline__ float fast_tanh(float x) {
    float ax = __builtin_fabsf(x);
    float e  = __builtin_amdgcn_exp2f(ax * (2.0f * LOG2E));
    float t  = 1.0f - 2.0f * __builtin_amdgcn_rcpf(e + 1.0f);
    return __builtin_copysignf(t, x);
}

__device__ __forceinline__ void copy16(const unsigned short* __restrict__ g,
                                       unsigned short* l) {
#if USE_ASYNC
    __builtin_amdgcn_global_load_async_to_lds_b128(
        (gptr_v4i)(void*)g, (lptr_v4i)(void*)l, 0, 0);
#else
    *(uint4*)l = *(const uint4*)g;
#endif
}

// ---------------- prepack kernels (one-time, bandwidth-bound) ----------------

// Ap[row][k] bf16, k = [x(0..1023) | h(1024..2047)], row-major
__global__ __launch_bounds__(256)
void pack_A_kernel(const float* __restrict__ x, const float* __restrict__ h,
                   unsigned short* __restrict__ Ap) {
    size_t t    = (size_t)blockIdx.x * 256 + threadIdx.x;
    size_t base = t * 8;                       // 8 elems per thread
    size_t row  = base >> 11;                  // / 2048
    int    k    = (int)(base & 2047);
    const float* src = (k < 1024) ? (x + row * 1024 + k)
                                  : (h + row * 1024 + (k - 1024));
    float4 f0 = *(const float4*)src;
    float4 f1 = *(const float4*)(src + 4);
    union { unsigned short s[8]; uint4 q; } p;
    p.s[0] = f2bf(f0.x); p.s[1] = f2bf(f0.y);
    p.s[2] = f2bf(f0.z); p.s[3] = f2bf(f0.w);
    p.s[4] = f2bf(f1.x); p.s[5] = f2bf(f1.y);
    p.s[6] = f2bf(f1.z); p.s[7] = f2bf(f1.w);
    *(uint4*)(Ap + base) = p.q;
}

// Wp[n][k] bf16 (column-major weights): n in [0,4096), k = [Wx | Wh] rows
__global__ __launch_bounds__(256)
void pack_W_kernel(const float* __restrict__ Wx, const float* __restrict__ Wh,
                   unsigned short* __restrict__ Wp) {
    int n  = blockIdx.x * 64 + (threadIdx.x & 63);
    int k0 = blockIdx.y * 32 + ((threadIdx.x >> 6) << 3);   // 8-elem k chunk
    const float* src = (k0 < 1024) ? Wx : Wh;
    int kb = k0 & 1023;
    union { unsigned short s[8]; uint4 q; } p;
#pragma unroll
    for (int j = 0; j < 8; ++j)
        p.s[j] = f2bf(src[(size_t)(kb + j) * (4 * HDIM) + n]);
    *(uint4*)(Wp + (size_t)n * KTOT + k0) = p.q;
}

// ---------------- fused GEMM + LSTM kernel ----------------

__device__ __forceinline__ void stage_tile(const unsigned short* __restrict__ Ap,
                                           const unsigned short* __restrict__ Wp,
                                           unsigned short* As, unsigned short* Bs,
                                           int tid, int m0, int n0, int k0) {
    // A tile: 256 rows x 32 k = 1024 x 16B chunks -> 4 per thread
#pragma unroll
    for (int i = 0; i < 4; ++i) {
        int ch  = tid + (i << 8);
        int row = ch >> 2;
        int ko  = (ch & 3) << 3;
        copy16(Ap + (size_t)(m0 + row) * KTOT + k0 + ko, As + row * LDA + ko);
    }
    // B tile: 128 cols (4 gates x 32) x 32 k -> 2 chunks per thread
#pragma unroll
    for (int i = 0; i < 2; ++i) {
        int ch   = tid + (i << 8);
        int cl   = ch >> 2;
        int ko   = (ch & 3) << 3;
        int ncol = (cl >> 5) * HDIM + n0 + (cl & 31);
        copy16(Wp + (size_t)ncol * KTOT + k0 + ko, Bs + cl * LDB + ko);
    }
}

// One k-step: 16 WMMAs in two half-batches (4 cached B frags each) to keep
// live registers inside the 256-VGPR direct-address window.
__device__ __forceinline__ void mma_step(const unsigned short* __restrict__ Asb,
                                         const unsigned short* __restrict__ Bsb,
                                         v8f (&acc)[2][2][4],
                                         int wave, int sel, int ln) {
    FragU a[2];
#pragma unroll
    for (int rb = 0; rb < 2; ++rb) {
        const unsigned short* ap = Asb + (wave * 32 + rb * 16 + ln) * LDA + (sel << 3);
        a[rb].q[0] = *(const uint4*)ap;
        a[rb].q[1] = *(const uint4*)(ap + 16);
    }
#pragma unroll
    for (int nb = 0; nb < 2; ++nb) {
        FragU b[4];
#pragma unroll
        for (int g = 0; g < 4; ++g) {
            const unsigned short* bp =
                Bsb + (g * NT + nb * 16 + ln) * LDB + (sel << 3);
            b[g].q[0] = *(const uint4*)bp;
            b[g].q[1] = *(const uint4*)(bp + 16);
        }
#pragma unroll
        for (int rb = 0; rb < 2; ++rb)
#pragma unroll
            for (int g = 0; g < 4; ++g)
                acc[rb][nb][g] = __builtin_amdgcn_wmma_f32_16x16x32_bf16(
                    false, a[rb].v, false, b[g].v,
                    (short)0, acc[rb][nb][g], false, false);
    }
}

__global__ __launch_bounds__(256, 1)
void lstm_gemm_wmma(const unsigned short* __restrict__ Ap,
                    const unsigned short* __restrict__ Wp,
                    const float* __restrict__ cin,
                    const float* __restrict__ bx,
                    float* __restrict__ out)
{
    __shared__ unsigned short As[2][MT * LDA];       // 2 x 20 KB
    __shared__ unsigned short Bs[2][4 * NT * LDB];   // 2 x 10 KB

    const int tid  = threadIdx.x;
    const int wave = tid >> 5;
    const int lane = tid & 31;
    const int sel  = lane >> 4;
    const int ln   = lane & 15;

    const int n0 = blockIdx.x * NT;
    const int m0 = blockIdx.y * MT;

    v8f acc[2][2][4];   // [row-block][n-block][gate]
#pragma unroll
    for (int rb = 0; rb < 2; ++rb)
#pragma unroll
        for (int nb = 0; nb < 2; ++nb)
#pragma unroll
            for (int g = 0; g < 4; ++g)
                acc[rb][nb][g] = (v8f)(0.0f);

    // prologue: stage first tile (6 async copies per thread)
    stage_tile(Ap, Wp, As[0], Bs[0], tid, m0, n0, 0);

    // steady state: unconditional stage-next + wait(6) — last iteration peeled.
    // unroll(disable): one loop body keeps only one iteration's fragments live,
    // avoiding spill into the VGPR-MSB region (s_set_vgpr_msb + mov churn).
#pragma clang loop unroll(disable)
    for (int k0 = 0; k0 < KTOT - KT; k0 += KT) {
        const int buf = (k0 >> 5) & 1;
        stage_tile(Ap, Wp, As[buf ^ 1], Bs[buf ^ 1], tid, m0, n0, k0 + KT);
#if USE_ASYNC
        asm volatile("s_wait_asynccnt 0x6" ::: "memory");
#endif
        __syncthreads();
        mma_step(As[buf], Bs[buf], acc, wave, sel, ln);
        __syncthreads();
    }

    // final tile (buf index of last k-step: (KTOT/KT - 1) & 1 == 1)
#if USE_ASYNC
    asm volatile("s_wait_asynccnt 0x0" ::: "memory");
#endif
    __syncthreads();
    mma_step(As[1], Bs[1], acc, wave, sel, ln);

    // ---- fused LSTM epilogue ----
    float* __restrict__ hout = out;
    float* __restrict__ cout = out + (size_t)BDIM * HDIM;
#pragma unroll
    for (int nb = 0; nb < 2; ++nb) {
        const int col = n0 + nb * 16 + ln;
        const float bi  = bx[col];
        const float bff = bx[HDIM + col];
        const float bgg = bx[2 * HDIM + col];
        const float bo  = bx[3 * HDIM + col];
#pragma unroll
        for (int rb = 0; rb < 2; ++rb) {
#pragma unroll
            for (int r = 0; r < 8; ++r) {
                const int row = m0 + wave * 32 + rb * 16 + sel * 8 + r;
                const float gi = acc[rb][nb][0][r] + bi;
                const float gf = acc[rb][nb][1][r] + bff;
                const float gg = acc[rb][nb][2][r] + bgg;
                const float go = acc[rb][nb][3][r] + bo;
                const float iv = fast_sigmoid(gi);
                const float fv = fast_sigmoid(gf);
                const float gv = fast_tanh(gg);
                const float ov = fast_sigmoid(go);
                const size_t idx = (size_t)row * HDIM + col;
                const float cn = fv * cin[idx] + iv * gv;
                cout[idx] = cn;
                hout[idx] = ov * fast_tanh(cn);
            }
        }
    }
}

extern "C" void kernel_launch(void* const* d_in, const int* in_sizes, int n_in,
                              void* d_out, int out_size, void* d_ws, size_t ws_size,
                              hipStream_t stream) {
    const float* x  = (const float*)d_in[0];
    const float* h  = (const float*)d_in[1];
    const float* c  = (const float*)d_in[2];
    const float* Wx = (const float*)d_in[3];
    const float* bx = (const float*)d_in[4];
    const float* Wh = (const float*)d_in[5];
    float* out = (float*)d_out;

    // workspace: Ap = 8192*2048 bf16 (32 MiB), Wp = 4096*2048 bf16 (16 MiB)
    unsigned short* Ap = (unsigned short*)d_ws;
    unsigned short* Wp = Ap + (size_t)BDIM * KTOT;

    pack_A_kernel<<<dim3(BDIM * KTOT / 8 / 256), dim3(256), 0, stream>>>(x, h, Ap);
    pack_W_kernel<<<dim3(4 * HDIM / 64, KTOT / 32), dim3(256), 0, stream>>>(Wx, Wh, Wp);

    dim3 grid(HDIM / NT, BDIM / MT);   // (32, 32)
    lstm_gemm_wmma<<<grid, dim3(256), 0, stream>>>(Ap, Wp, c, bx, out);
}
